// AttentionBlock_33947421508019
// MI455X (gfx1250) — matrix operand order repaired and checked by reading, not compiled, not run on hardware
//
#include <hip/hip_runtime.h>
#include <hip/hip_bf16.h>

// ---------------------------------------------------------------------------
// AttentionBlock on MI455X (gfx1250, wave32, WMMA f16 16x16x32 + TDM).
// b=4, c=256, h=w=64 -> s=4096, heads=2, hd=128, groups=8.
// ---------------------------------------------------------------------------

#define Cc   256
#define Ss   4096
#define HD   128
#define CPG  32     // channels per group (256/8)

typedef __attribute__((ext_vector_type(16))) _Float16 v16h;
typedef __attribute__((ext_vector_type(8)))  _Float16 v8h;
typedef __attribute__((ext_vector_type(8)))  float    v8f;
typedef __attribute__((ext_vector_type(4)))  unsigned int v4u;
typedef __attribute__((ext_vector_type(8)))  int      v8i;
typedef __attribute__((ext_vector_type(4)))  int      v4i;

union V16 { v16h v; v8h h[2]; };

#if defined(__has_builtin)
#if __has_builtin(__builtin_amdgcn_tensor_load_to_lds) && \
    __has_builtin(__builtin_amdgcn_s_wait_tensorcnt)
#define USE_TDM 1
#endif
#endif
#ifndef USE_TDM
#define USE_TDM 0
#endif

// WMMA wrapper: D = A(16x32 f16) * B(32x16 f16) + C(16x16 f32)
static __device__ __forceinline__ v8f wmma16(v16h a, v16h b, v8f c) {
  return __builtin_amdgcn_wmma_f32_16x16x32_f16(
      /*neg_a=*/false, a, /*neg_b=*/false, b,
      /*c_mod=*/(short)0, c, /*reuse_a=*/false, /*reuse_b=*/false);
}

// A fragment (16x32, MxK). rowbase -> element [m][0] of row-major [M][Kstride]
// for this lane's row m = lane&15. ISA layout: halves 0..7 = K=kbase+0..7,
// halves 8..15 = K=16+kbase+0..7, kbase=(lane>>4)*8 -> two 16B loads.
static __device__ __forceinline__ v16h load_a_frag(const _Float16* rowbase,
                                                   int kk, int lane) {
  int kbase = (lane >> 4) * 8;
  V16 u;
  u.h[0] = *(const v8h*)(rowbase + kk + kbase);
  u.h[1] = *(const v8h*)(rowbase + kk + 16 + kbase);
  return u.v;
}

// B fragment (32x16, KxN). colbase -> element [n][0] of row-major [N][Kstride]
// for this lane's column n = lane&15. Same K order as A: halves 0..7 =
// K=kb+0..7, halves 8..15 = K=16+kb+0..7, kb=(lane>>4)*8 (two 16B loads).
static __device__ __forceinline__ v16h load_b_frag(const _Float16* colbase,
                                                   int kk, int lane) {
  int kb = (lane >> 4) * 8;
  V16 u;
  u.h[0] = *(const v8h*)(colbase + kk + kb);
  u.h[1] = *(const v8h*)(colbase + kk + 16 + kb);
  return u.v;
}

#if USE_TDM
// Tensor Data Mover: async DMA of a [tile_rows x tile_cols] f16 tile from a
// row-major global matrix (row stride = stride_elems) into LDS at lds_off.
// D# fields packed per CDNA5 ISA 8.3/8.4 (group0: count=1,type=2; group1:
// data_size=1(2B), tensor_dim0=stride, tensor_dim1=1<<20, tile dims, stride).
// This toolchain's builtin takes 6 args: (g0 v4u, g1 v8i, g2 v4i, g3 v4i,
// g4 v8i, cpol) — g4 zeroed (2D tile needs only groups 0/1).
static __device__ __forceinline__ void tdm_load_tile(unsigned lds_off,
                                                     const void* gptr,
                                                     unsigned tile_rows,
                                                     unsigned tile_cols,
                                                     unsigned stride_elems) {
  unsigned long long ga = (unsigned long long)(size_t)gptr;
  v4u g0;
  g0[0] = 1u;                                          // count=1, is_restore=0
  g0[1] = lds_off;                                     // lds_addr (bytes)
  g0[2] = (unsigned)(ga & 0xffffffffu);                // global_addr[31:0]
  g0[3] = (unsigned)((ga >> 32) & 0x01ffffffu) | (2u << 30); // [56:32] | type=2
  const unsigned td1 = 1u << 20;                       // generous dim1 bound
  v8i g1;
  g1[0] = (int)(1u << 16);                             // data_size = 2 bytes
  g1[1] = (int)((stride_elems & 0xffffu) << 16);       // tensor_dim0 lo16
  g1[2] = (int)(((stride_elems >> 16) & 0xffffu) | ((td1 & 0xffffu) << 16));
  g1[3] = (int)(((td1 >> 16) & 0xffffu) | ((tile_cols & 0xffffu) << 16));
  g1[4] = (int)(tile_rows & 0xffffu);                  // tile_dim1 (tile_dim2=0)
  g1[5] = (int)stride_elems;                           // tensor_dim0_stride lo32
  g1[6] = 0;
  g1[7] = 0;
  v4i g2 = {0, 0, 0, 0};                               // 2D tile: groups 2/3 idle
  v4i g3 = {0, 0, 0, 0};
  v8i g4 = {0, 0, 0, 0, 0, 0, 0, 0};                   // unused trailing group
  __builtin_amdgcn_tensor_load_to_lds(g0, g1, g2, g3, g4, 0);
}

static __device__ __forceinline__ unsigned lds_off_of(const void* p) {
  // LDS aperture: low 32 bits of a generic LDS address are the LDS byte offset.
  return (unsigned)(size_t)p;
}
#endif

// ---------------------------------------------------------------------------
// Kernel 0: f32 -> f16 weight conversion.
// ---------------------------------------------------------------------------
__global__ void f32_to_f16_kernel(const float* __restrict__ src,
                                  _Float16* __restrict__ dst, int n) {
  int i = blockIdx.x * blockDim.x + threadIdx.x;
  if (i < n) dst[i] = (_Float16)src[i];
}

// ---------------------------------------------------------------------------
// Kernel 1: GroupNorm, writes x_norm transposed as f16 [b][s][c].
// ---------------------------------------------------------------------------
__global__ void groupnorm_kernel(const float* __restrict__ x,
                                 const float* __restrict__ gamma,
                                 const float* __restrict__ beta,
                                 _Float16* __restrict__ xn_t) {
  __shared__ float ssum[256];
  __shared__ float ssq[256];
  const int bg  = blockIdx.x;          // b*8 + g
  const int b   = bg >> 3;
  const int g   = bg & 7;
  const int tid = threadIdx.x;
  const float* base = x + ((size_t)b * Cc + (size_t)g * CPG) * Ss;

  float sum = 0.f, sq = 0.f;
  for (int i = tid; i < CPG * Ss; i += 256) {
    float v = base[i];
    sum += v;
    sq  += v * v;
  }
  ssum[tid] = sum;
  ssq[tid]  = sq;
  __syncthreads();
  for (int st = 128; st > 0; st >>= 1) {
    if (tid < st) { ssum[tid] += ssum[tid + st]; ssq[tid] += ssq[tid + st]; }
    __syncthreads();
  }
  const float inv_n = 1.0f / (float)(CPG * Ss);
  const float mean  = ssum[0] * inv_n;
  const float var   = ssq[0] * inv_n - mean * mean;
  const float rstd  = rsqrtf(var + 1e-5f);

  for (int i = tid; i < CPG * Ss; i += 256) {
    int c = g * CPG + i / Ss;
    int p = i % Ss;
    float v = (base[i] - mean) * rstd * gamma[c] + beta[c];
    xn_t[((size_t)b * Ss + p) * Cc + c] = (_Float16)v;
  }
}

// ---------------------------------------------------------------------------
// Shared GEMM body pieces: 128x128 block tile, 8 waves of 32x64, K-step 32.
// A/B 128x32 tiles stream through double-buffered LDS via the TDM, with
// s_wait_tensorcnt(2) pipelining the next pair against current WMMAs.
// ---------------------------------------------------------------------------
#define GEMM_TILE_LOOP(Abase, Bbase, STRIDE)                                   \
  const int NIT = (STRIDE) / 32;                                               \
  _Pragma("unroll 1")                                                          \
  for (int it = 0; it < NIT; ++it) {                                           \
    const int cur = it & 1;                                                    \
    __syncthreads(); /* prior iteration done reading buf cur^1 */              \
    TILE_FILL(Abase, Bbase, STRIDE)                                            \
    __syncthreads(); /* buf cur ready for everyone */                          \
    v16h a0 = load_a_frag(&At[cur][(wm * 32 + m) * 32], 0, lane);              \
    v16h a1 = load_a_frag(&At[cur][(wm * 32 + 16 + m) * 32], 0, lane);         \
    _Pragma("unroll")                                                          \
    for (int nt = 0; nt < 4; ++nt) {                                           \
      v16h bb = load_b_frag(&Bt[cur][(wn * 64 + nt * 16 + m) * 32], 0, lane);  \
      acc[0][nt] = wmma16(a0, bb, acc[0][nt]);                                 \
      acc[1][nt] = wmma16(a1, bb, acc[1][nt]);                                 \
    }                                                                          \
  }

#if USE_TDM
#define TILE_PRIME(Abase, Bbase, STRIDE)                                       \
  if (wave == 0) {                                                             \
    tdm_load_tile(lds_off_of(&At[0][0]), (Abase), 128, 32, (STRIDE));          \
    tdm_load_tile(lds_off_of(&Bt[0][0]), (Bbase), 128, 32, (STRIDE));          \
  }
#define TILE_FILL(Abase, Bbase, STRIDE)                                        \
  if (wave == 0) {                                                             \
    if (it + 1 < NIT) {                                                        \
      tdm_load_tile(lds_off_of(&At[cur ^ 1][0]),                               \
                    (Abase) + (it + 1) * 32, 128, 32, (STRIDE));               \
      tdm_load_tile(lds_off_of(&Bt[cur ^ 1][0]),                               \
                    (Bbase) + (it + 1) * 32, 128, 32, (STRIDE));               \
      __builtin_amdgcn_s_wait_tensorcnt(2); /* pair for buf cur complete */    \
    } else {                                                                   \
      __builtin_amdgcn_s_wait_tensorcnt(0);                                    \
    }                                                                          \
  }
#else
#define TILE_PRIME(Abase, Bbase, STRIDE)
#define TILE_FILL(Abase, Bbase, STRIDE)                                        \
  for (int i = tid; i < 512; i += 256) {                                       \
    int row = i >> 2, cp = i & 3;                                              \
    *(v8h*)&At[cur][row * 32 + cp * 8] =                                       \
        *(const v8h*)((Abase) + (size_t)row * (STRIDE) + it * 32 + cp * 8);    \
    *(v8h*)&Bt[cur][row * 32 + cp * 8] =                                       \
        *(const v8h*)((Bbase) + (size_t)row * (STRIDE) + it * 32 + cp * 8);    \
  }
#endif

// ---------------------------------------------------------------------------
// Kernel 2: QKV GEMM. C[o, s] = W[o, c] * Xn_t[s, c]^T + bias. Scatters:
//   q (scaled) -> [bh][s][hd], k -> [bh][s][hd], v -> [bh][hd][s]  (f16)
// ---------------------------------------------------------------------------
__global__ void qkv_gemm_kernel(const _Float16* __restrict__ W,    // [768][256]
                                const _Float16* __restrict__ Xn,   // [b][S][C]
                                const float*    __restrict__ bias, // [768]
                                _Float16* __restrict__ qt,
                                _Float16* __restrict__ kt,
                                _Float16* __restrict__ vv) {
  __shared__ __align__(16) _Float16 At[2][128 * 32];
  __shared__ __align__(16) _Float16 Bt[2][128 * 32];
  const int b    = blockIdx.z;
  const int Mb   = blockIdx.y * 128;
  const int Nb   = blockIdx.x * 128;
  const int tid  = threadIdx.x;
  const int wave = tid >> 5;
  const int lane = tid & 31;
  const int wm   = wave >> 1;          // 0..3
  const int wn   = wave & 1;           // 0..1
  const int m    = lane & 15;
  const _Float16* Abase = W + (size_t)Mb * Cc;
  const _Float16* Bbase = Xn + (size_t)b * Ss * Cc + (size_t)Nb * Cc;

  v8f acc[2][4] = {};
  TILE_PRIME(Abase, Bbase, Cc)
  GEMM_TILE_LOOP(Abase, Bbase, Cc)

  const float scale = 0.088388347648318447f;   // 1/sqrt(128)
#pragma unroll
  for (int mt = 0; mt < 2; ++mt) {
#pragma unroll
    for (int nt = 0; nt < 4; ++nt) {
#pragma unroll
      for (int r = 0; r < 8; ++r) {
        int o  = Mb + wm * 32 + mt * 16 + r + (lane >> 4) * 8;
        int sg = Nb + wn * 64 + nt * 16 + (lane & 15);
        float val = acc[mt][nt][r] + bias[o];
        int which = o >> 8;          // 0=q, 1=k, 2=v
        int head  = (o >> 7) & 1;
        int d     = o & 127;
        size_t bh = (size_t)b * 2 + head;
        if (which == 0)
          qt[(bh * Ss + sg) * HD + d] = (_Float16)(val * scale);
        else if (which == 1)
          kt[(bh * Ss + sg) * HD + d] = (_Float16)val;
        else
          vv[(bh * HD + d) * Ss + sg] = (_Float16)val;
      }
    }
  }
}

// ---------------------------------------------------------------------------
// Kernel 3: flash attention. One block per (s-tile of 128, bh). Each wave
// owns 16 query rows, streams keys/values in chunks of 32, online softmax.
// ---------------------------------------------------------------------------
__global__ void attn_kernel(const _Float16* __restrict__ qt,  // [bh][S][HD]
                            const _Float16* __restrict__ kt,  // [bh][S][HD]
                            const _Float16* __restrict__ vv,  // [bh][HD][S]
                            _Float16* __restrict__ ao) {      // [b][S][C]
  __shared__ __align__(16) _Float16 Plds[8][16 * 32];
  const int bh   = blockIdx.y;        // b*2 + head
  const int b    = bh >> 1;
  const int head = bh & 1;
  const int tid  = threadIdx.x;
  const int wave = tid >> 5;
  const int lane = tid & 31;
  const int m    = lane & 15;
  const int qbase = blockIdx.x * 128 + wave * 16;

  const _Float16* Q = qt + (size_t)bh * Ss * HD;
  const _Float16* K = kt + (size_t)bh * Ss * HD;
  const _Float16* V = vv + (size_t)bh * HD * Ss;

  v16h aq[4];
#pragma unroll
  for (int j = 0; j < 4; ++j)
    aq[j] = load_a_frag(Q + (size_t)(qbase + m) * HD, j * 32, lane);

  v8f O[8] = {};
  float rowm[8], rowl[8];
#pragma unroll
  for (int r = 0; r < 8; ++r) { rowm[r] = -__builtin_inff(); rowl[r] = 0.f; }

  for (int t0 = 0; t0 < Ss; t0 += 32) {
    // Near-cache prefetch of the next key/value chunk.
    __builtin_prefetch(K + (size_t)(t0 + 32 + m) * HD, 0, 3);
    __builtin_prefetch(V + (size_t)m * Ss + t0 + 32, 0, 3);

    // S = Q^T K for 16 rows x 32 key columns (two 16x16 tiles).
    v8f c0 = {}, c1 = {};
#pragma unroll
    for (int j = 0; j < 4; ++j) {
      v16h bk0 = load_b_frag(K + (size_t)(t0 + m) * HD,      j * 32, lane);
      v16h bk1 = load_b_frag(K + (size_t)(t0 + 16 + m) * HD, j * 32, lane);
      c0 = wmma16(aq[j], bk0, c0);
      c1 = wmma16(aq[j], bk1, c1);
    }

    // Online softmax over C-fragment rows (VGPR r <-> row r / r+8).
#pragma unroll
    for (int r = 0; r < 8; ++r) {
      float cand = fmaxf(c0[r], c1[r]);
#pragma unroll
      for (int off = 1; off < 16; off <<= 1)
        cand = fmaxf(cand, __shfl_xor(cand, off, 16));
      float newm = fmaxf(rowm[r], cand);
      float sc   = __expf(rowm[r] - newm);
      float p0   = __expf(c0[r] - newm);
      float p1   = __expf(c1[r] - newm);
      float ps   = p0 + p1;
#pragma unroll
      for (int off = 1; off < 16; off <<= 1)
        ps += __shfl_xor(ps, off, 16);
      rowl[r] = rowl[r] * sc + ps;
      rowm[r] = newm;
#pragma unroll
      for (int dt = 0; dt < 8; ++dt) O[dt][r] *= sc;
      int mrow = r + ((lane >> 4) << 3);
      Plds[wave][mrow * 32 + (lane & 15)]      = (_Float16)p0;
      Plds[wave][mrow * 32 + 16 + (lane & 15)] = (_Float16)p1;
    }

    // Re-layout P (C-frag in LDS) as an A-fragment (same-wave DS is in-order).
    v16h pf;
    {
      int kbase = (lane >> 4) * 8;
      const _Float16* pb = &Plds[wave][m * 32];
      V16 u;
      u.h[0] = *(const v8h*)(pb + kbase);
      u.h[1] = *(const v8h*)(pb + 16 + kbase);
      pf = u.v;
    }

    // O[16x128] += P[16x32] * V^T[32x128].
#pragma unroll
    for (int dt = 0; dt < 8; ++dt) {
      v16h bv = load_b_frag(V + (size_t)(dt * 16 + m) * Ss, t0, lane);
      O[dt] = wmma16(pf, bv, O[dt]);
    }
  }

#pragma unroll
  for (int r = 0; r < 8; ++r) {
    float inv = 1.0f / rowl[r];
    int sg = qbase + r + (lane >> 4) * 8;
#pragma unroll
    for (int dt = 0; dt < 8; ++dt) {
      int c = head * HD + dt * 16 + (lane & 15);
      ao[((size_t)b * Ss + sg) * Cc + c] = (_Float16)(O[dt][r] * inv);
    }
  }
}

// ---------------------------------------------------------------------------
// Kernel 4: projection GEMM + bias + residual (f32 out).
// ---------------------------------------------------------------------------
__global__ void proj_kernel(const _Float16* __restrict__ W,    // [256][256]
                            const _Float16* __restrict__ AO,   // [b][S][C]
                            const float*    __restrict__ bias, // [256]
                            const float*    __restrict__ x,    // [b][C][S]
                            float*          __restrict__ out) {
  __shared__ __align__(16) _Float16 At[2][128 * 32];
  __shared__ __align__(16) _Float16 Bt[2][128 * 32];
  const int b    = blockIdx.z;
  const int Mb   = blockIdx.y * 128;
  const int Nb   = blockIdx.x * 128;
  const int tid  = threadIdx.x;
  const int wave = tid >> 5;
  const int lane = tid & 31;
  const int wm   = wave >> 1;
  const int wn   = wave & 1;
  const int m    = lane & 15;
  const _Float16* Abase = W + (size_t)Mb * Cc;
  const _Float16* Bbase = AO + (size_t)b * Ss * Cc + (size_t)Nb * Cc;

  v8f acc[2][4] = {};
  TILE_PRIME(Abase, Bbase, Cc)
  GEMM_TILE_LOOP(Abase, Bbase, Cc)

#pragma unroll
  for (int mt = 0; mt < 2; ++mt) {
#pragma unroll
    for (int nt = 0; nt < 4; ++nt) {
#pragma unroll
      for (int r = 0; r < 8; ++r) {
        int o  = Mb + wm * 32 + mt * 16 + r + (lane >> 4) * 8;
        int sg = Nb + wn * 64 + nt * 16 + (lane & 15);
        size_t idx = ((size_t)b * Cc + o) * Ss + sg;
        out[idx] = acc[mt][nt][r] + bias[o] + x[idx];
      }
    }
  }
}

// ---------------------------------------------------------------------------
// Host launcher.
// ---------------------------------------------------------------------------
extern "C" void kernel_launch(void* const* d_in, const int* in_sizes, int n_in,
                              void* d_out, int out_size, void* d_ws, size_t ws_size,
                              hipStream_t stream) {
  const float* x      = (const float*)d_in[0];
  const float* gamma  = (const float*)d_in[1];
  const float* beta   = (const float*)d_in[2];
  const float* qkv_w  = (const float*)d_in[3];
  const float* qkv_b  = (const float*)d_in[4];
  const float* proj_w = (const float*)d_in[5];
  const float* proj_b = (const float*)d_in[6];
  float* out = (float*)d_out;

  char* ws = (char*)d_ws;
  const size_t MB = 1024ull * 1024ull;
  _Float16* xn_t = (_Float16*)(ws + 0 * MB);   // [b][S][C]     8 MB
  _Float16* qt   = (_Float16*)(ws + 8 * MB);   // [bh][S][HD]   8 MB
  _Float16* kt   = (_Float16*)(ws + 16 * MB);  // [bh][S][HD]   8 MB
  _Float16* vv   = (_Float16*)(ws + 24 * MB);  // [bh][HD][S]   8 MB
  _Float16* ao   = (_Float16*)(ws + 32 * MB);  // [b][S][C]     8 MB
  _Float16* qW16 = (_Float16*)(ws + 40 * MB);  // 768*256       0.375 MB
  _Float16* pW16 = (_Float16*)(ws + 40 * MB + 768 * 256 * sizeof(_Float16));

  f32_to_f16_kernel<<<(768 * 256 + 255) / 256, 256, 0, stream>>>(qkv_w, qW16, 768 * 256);
  f32_to_f16_kernel<<<(256 * 256 + 255) / 256, 256, 0, stream>>>(proj_w, pW16, 256 * 256);

  groupnorm_kernel<<<32, 256, 0, stream>>>(x, gamma, beta, xn_t);

  qkv_gemm_kernel<<<dim3(32, 6, 4), 256, 0, stream>>>(qW16, xn_t, qkv_b, qt, kt, vv);

  attn_kernel<<<dim3(32, 8), 256, 0, stream>>>(qt, kt, vv, ao);

  proj_kernel<<<dim3(32, 2, 4), 256, 0, stream>>>(pW16, ao, proj_b, x, out);
}
